// TemporalAwareCrossAttention_83597243449588
// MI455X (gfx1250) — matrix operand order repaired and checked by reading, not compiled, not run on hardware
//
#include <hip/hip_runtime.h>
#include <hip/hip_bf16.h>
#include <math.h>

#define DIM       512
#define HEADS     16
#define DIM_HEAD  64
#define INNER     1024
#define B_        2
#define T_        16
#define S_        128
#define NTOK      2048   // T_*S_

typedef __attribute__((ext_vector_type(16))) __bf16 bf16x16;
typedef __attribute__((ext_vector_type(8)))  float  floatx8;

struct alignas(16) U4 { unsigned int x, y, z, w; };
union AB { bf16x16 v; U4 q[2]; };
union PK8 { U4 q; unsigned short s[8]; };

static __device__ __forceinline__ unsigned short f2bf(float f) {
    union { float f; unsigned int u; } v; v.f = f;
    unsigned int r = v.u + 0x7FFFu + ((v.u >> 16) & 1u);   // round-to-nearest-even
    return (unsigned short)(r >> 16);
}

#define WMMA_BF16(A, Bm, C) __builtin_amdgcn_wmma_f32_16x16x32_bf16( \
    false, (A).v, false, (Bm).v, (short)0, (C), false, false)

// ---------------- prep kernels ----------------

__global__ __launch_bounds__(256)
void k_addpos_bf16(const float* __restrict__ src, const float* __restrict__ pos,
                   unsigned short* __restrict__ dst, int n) {
    int i = blockIdx.x * 256 + threadIdx.x;
    if (i >= n) return;
    int d = i & (DIM - 1);
    int t = (i / (S_ * DIM)) & (T_ - 1);
    dst[i] = f2bf(src[i] + pos[t * DIM + d]);
}

// WT[c*rows + r] = bf16(W[r*cols + c])   (N-major weights for WMMA B operands)
__global__ __launch_bounds__(256)
void k_transpose_bf16(const float* __restrict__ W, unsigned short* __restrict__ WT,
                      int rows, int cols) {
    int i = blockIdx.x * 256 + threadIdx.x;
    if (i >= rows * cols) return;
    int r = i / cols, c = i % cols;
    WT[c * rows + r] = f2bf(W[i]);
}

// ---------------- QKV projection + RoPE ----------------
// one wave computes a 32-row x 64-col (one head) tile of Q, K or V,
// with double-buffered operand loads to overlap VMEM with the matrix pipe.

__global__ __launch_bounds__(128)
void k_qkv(const unsigned short* __restrict__ xb, const unsigned short* __restrict__ yb,
           const unsigned short* __restrict__ WqT, const unsigned short* __restrict__ WkT,
           const unsigned short* __restrict__ WvT,
           unsigned short* __restrict__ qb, unsigned short* __restrict__ kb,
           unsigned short* __restrict__ vt) {
    const int lane = threadIdx.x & 31;
    const int lr = lane & 15, hl = lane >> 4;
    int task = blockIdx.x * 4 + (threadIdx.x >> 5);      // 6144 tasks
    int mat = task / 2048;                               // 0=Q 1=K 2=V
    int rem = task % 2048;
    int b   = rem / 1024; rem %= 1024;
    int rt  = rem >> 4;                                  // 64 tiles of 32 rows
    int cg  = rem & 15;                                  // head / 64-col group

    const unsigned short* A  = (mat == 2 ? yb : xb) + (size_t)b * NTOK * DIM;
    const unsigned short* Wt = (mat == 0 ? WqT : (mat == 1 ? WkT : WvT));

    floatx8 zf = {0.f,0.f,0.f,0.f,0.f,0.f,0.f,0.f};
    floatx8 acc[2][4];
    #pragma unroll
    for (int s = 0; s < 2; ++s)
        #pragma unroll
        for (int t = 0; t < 4; ++t) acc[s][t] = zf;

    const unsigned short* ar0 = A + (size_t)(rt * 32 + lr) * DIM;
    const unsigned short* ar1 = A + (size_t)(rt * 32 + 16 + lr) * DIM;
    const unsigned short* bp0 = Wt + (size_t)(cg * 64 +  0 + lr) * DIM + hl * 16;
    const unsigned short* bp1 = Wt + (size_t)(cg * 64 + 16 + lr) * DIM + hl * 16;
    const unsigned short* bp2 = Wt + (size_t)(cg * 64 + 32 + lr) * DIM + hl * 16;
    const unsigned short* bp3 = Wt + (size_t)(cg * 64 + 48 + lr) * DIM + hl * 16;

    AB aBuf[2][2], bBuf[2][4];
#define QKV_LOAD(buf, ks) do {                                                \
        aBuf[buf][0].q[0] = *(const U4*)(ar0 + (ks)*32 +     hl * 16);        \
        aBuf[buf][0].q[1] = *(const U4*)(ar0 + (ks)*32 + 8 + hl * 16);        \
        aBuf[buf][1].q[0] = *(const U4*)(ar1 + (ks)*32 +     hl * 16);        \
        aBuf[buf][1].q[1] = *(const U4*)(ar1 + (ks)*32 + 8 + hl * 16);        \
        bBuf[buf][0].q[0] = *(const U4*)(bp0 + (ks)*32);                      \
        bBuf[buf][0].q[1] = *(const U4*)(bp0 + (ks)*32 + 8);                  \
        bBuf[buf][1].q[0] = *(const U4*)(bp1 + (ks)*32);                      \
        bBuf[buf][1].q[1] = *(const U4*)(bp1 + (ks)*32 + 8);                  \
        bBuf[buf][2].q[0] = *(const U4*)(bp2 + (ks)*32);                      \
        bBuf[buf][2].q[1] = *(const U4*)(bp2 + (ks)*32 + 8);                  \
        bBuf[buf][3].q[0] = *(const U4*)(bp3 + (ks)*32);                      \
        bBuf[buf][3].q[1] = *(const U4*)(bp3 + (ks)*32 + 8);                  \
    } while (0)

    QKV_LOAD(0, 0);
    #pragma unroll 2
    for (int ks = 0; ks < 16; ++ks) {
        int cur = ks & 1, nxt = cur ^ 1;
        if (ks < 15) QKV_LOAD(nxt, ks + 1);              // prefetch next K-step
        #pragma unroll
        for (int t = 0; t < 4; ++t) {
            acc[0][t] = WMMA_BF16(aBuf[cur][0], bBuf[cur][t], acc[0][t]);
            acc[1][t] = WMMA_BF16(aBuf[cur][1], bBuf[cur][t], acc[1][t]);
        }
    }
#undef QKV_LOAD

    if (mat < 2) {
        // RoPE: angle(ch) = p * 10000^(-(ch&31)/32); rotate_half pairs tiles (0,2),(1,3)
        unsigned short* dst = (mat == 0 ? qb : kb);
        const float L32 = 0.2878231366242557f;           // ln(10000)/32
        float invf_lo = __expf(-(float)lr * L32);
        float invf_hi = __expf(-(float)(lr + 16) * L32);
        size_t base = (size_t)(b * HEADS + cg) * NTOK * DIM_HEAD;
        #pragma unroll
        for (int s = 0; s < 2; ++s) {
            #pragma unroll
            for (int r = 0; r < 8; ++r) {
                int   p  = rt * 32 + s * 16 + r + 8 * hl;
                float pf = (float)p;
                float a0 = pf * invf_lo, a1 = pf * invf_hi;
                float c0 = __cosf(a0), s0 = __sinf(a0);
                float c1 = __cosf(a1), s1 = __sinf(a1);
                float o0 = acc[s][0][r], o1 = acc[s][1][r];
                float o2 = acc[s][2][r], o3 = acc[s][3][r];
                size_t rb = base + (size_t)p * DIM_HEAD;
                dst[rb +  0 + lr] = f2bf(o0 * c0 - o2 * s0);
                dst[rb + 16 + lr] = f2bf(o1 * c1 - o3 * s1);
                dst[rb + 32 + lr] = f2bf(o2 * c0 + o0 * s0);
                dst[rb + 48 + lr] = f2bf(o3 * c1 + o1 * s1);
            }
        }
    } else {
        // V stored transposed vt[b,h,dim,tok]; rows are consecutive per lane -> b128 stores
        size_t base = (size_t)(b * HEADS + cg) * DIM_HEAD * NTOK;
        #pragma unroll
        for (int s = 0; s < 2; ++s) {
            int p0 = rt * 32 + s * 16 + hl * 8;
            #pragma unroll
            for (int t = 0; t < 4; ++t) {
                PK8 pk;
                #pragma unroll
                for (int r = 0; r < 8; ++r) pk.s[r] = f2bf(acc[s][t][r]);
                *(U4*)(vt + base + (size_t)(t * 16 + lr) * NTOK + p0) = pk.q;
            }
        }
    }
}

// ---------------- flash attention (single wave per block) ----------------

__global__ __launch_bounds__(32)
void k_attn(const unsigned short* __restrict__ qb, const unsigned short* __restrict__ kb,
            const unsigned short* __restrict__ vt, const int* __restrict__ tmask,
            unsigned short* __restrict__ attn_bf) {
    __shared__ __align__(16) float          Sls[16 * 32];
    __shared__ __align__(16) unsigned short Pls[16 * 32];
    __shared__ float fLDS[16];
    __shared__ float lLDS[16];

    const int lane = threadIdx.x;
    const int lr = lane & 15, hl = lane >> 4;
    int task = blockIdx.x;                               // 4096 = B*H*128
    int b  = task / (HEADS * 128);
    int rem = task % (HEADS * 128);
    int h  = rem / 128;
    int qt = rem % 128;
    int q0 = qt * 16;
    int ti = qt >> 3;                                    // temporal row of this q tile

    const size_t bh = (size_t)(b * HEADS + h);
    const unsigned short* qbase = qb + bh * NTOK * DIM_HEAD;
    const unsigned short* kbase = kb + bh * NTOK * DIM_HEAD;
    const unsigned short* vbase = vt + bh * DIM_HEAD * NTOK;

    // temporal mask row -> bitmask (loaded once)
    const int* mrow = tmask + (b * T_ + ti) * T_;
    unsigned mbits = 0;
    #pragma unroll
    for (int tj = 0; tj < T_; ++tj) mbits |= (mrow[tj] != 0 ? 1u : 0u) << tj;

    // Q tile as two A operands (K = d 0..31 and 32..63)
    AB qA0, qA1;
    {
        const unsigned short* qp = qbase + (size_t)(q0 + lr) * DIM_HEAD;
        qA0.q[0] = *(const U4*)(qp +      hl * 16);
        qA0.q[1] = *(const U4*)(qp +  8 + hl * 16);
        qA1.q[0] = *(const U4*)(qp + 32 + hl * 16);
        qA1.q[1] = *(const U4*)(qp + 40 + hl * 16);
    }

    floatx8 zf = {0.f,0.f,0.f,0.f,0.f,0.f,0.f,0.f};
    floatx8 O[4];
    #pragma unroll
    for (int t = 0; t < 4; ++t) O[t] = zf;
    float m_run = -1e30f, l_run = 0.f;

    for (int j0 = 0; j0 < NTOK; j0 += 32) {
        if (!((mbits >> (j0 >> 7)) & 1u)) continue;      // masked temporal block: exact skip

        // S = Q @ K^T * scale  (two 16x16 tiles covering 32 keys)
        #pragma unroll
        for (int c = 0; c < 2; ++c) {
            AB kB0, kB1;
            const unsigned short* kp = kbase + (size_t)(j0 + c * 16 + lr) * DIM_HEAD + hl * 16;
            kB0.q[0] = *(const U4*)(kp);
            kB0.q[1] = *(const U4*)(kp + 8);
            kB1.q[0] = *(const U4*)(kp + 32);
            kB1.q[1] = *(const U4*)(kp + 40);
            floatx8 s = WMMA_BF16(qA0, kB0, zf);
            s = WMMA_BF16(qA1, kB1, s);
            #pragma unroll
            for (int r = 0; r < 8; ++r)
                Sls[(r + 8 * hl) * 32 + c * 16 + lr] = s[r] * 0.125f;  // DIM_HEAD^-0.5
        }

        // prefetch V operands now: these loads overlap the LDS softmax below
        AB vB[4];
        #pragma unroll
        for (int t = 0; t < 4; ++t) {
            const unsigned short* vp = vbase + (size_t)(t * 16 + lr) * NTOK + j0 + hl * 16;
            vB[t].q[0] = *(const U4*)(vp);
            vB[t].q[1] = *(const U4*)(vp + 8);
        }
        __syncthreads();

        // online softmax stats: lane owns row lr (both half-waves write identical values)
        {
            float bm = -1e30f;
            #pragma unroll
            for (int cc = 0; cc < 32; ++cc) bm = fmaxf(bm, Sls[lr * 32 + cc]);
            float nm  = fmaxf(m_run, bm);
            float fac = __expf(m_run - nm);
            float ps  = 0.f;
            #pragma unroll
            for (int cc = 0; cc < 32; ++cc) {
                float pv = __expf(Sls[lr * 32 + cc] - nm);
                ps += pv;
                Pls[lr * 32 + cc] = f2bf(pv);
            }
            l_run = l_run * fac + ps;
            m_run = nm;
            fLDS[lr] = fac;
        }
        __syncthreads();

        // rescale running output accumulators
        float f8[8];
        #pragma unroll
        for (int r = 0; r < 8; ++r) f8[r] = fLDS[hl * 8 + r];
        #pragma unroll
        for (int t = 0; t < 4; ++t)
            #pragma unroll
            for (int r = 0; r < 8; ++r) O[t][r] *= f8[r];

        // P as A operand from LDS; V already in registers
        AB pA;
        pA.q[0] = *(const U4*)(&Pls[lr * 32 +     hl * 16]);
        pA.q[1] = *(const U4*)(&Pls[lr * 32 + 8 + hl * 16]);
        #pragma unroll
        for (int t = 0; t < 4; ++t)
            O[t] = WMMA_BF16(pA, vB[t], O[t]);
        __syncthreads();
    }

    lLDS[lr] = l_run;
    __syncthreads();
    float inv8[8];
    #pragma unroll
    for (int r = 0; r < 8; ++r) inv8[r] = 1.0f / lLDS[hl * 8 + r];

    // store head output as bf16 [b, tok, h*64+ch] for the final GEMM
    #pragma unroll
    for (int r = 0; r < 8; ++r) {
        size_t row = (size_t)b * NTOK + q0 + r + 8 * hl;
        #pragma unroll
        for (int t = 0; t < 4; ++t)
            attn_bf[row * INNER + h * 64 + t * 16 + lr] = f2bf(O[t][r] * inv8[r]);
    }
}

// ---------------- output projection: proj = attn @ Wo + bo ----------------
// one wave computes a 16-row x 64-col tile, double-buffered K-loop

__global__ __launch_bounds__(128)
void k_proj(const unsigned short* __restrict__ attn_bf, const unsigned short* __restrict__ WoT,
            const float* __restrict__ bo, float* __restrict__ proj) {
    const int lane = threadIdx.x & 31;
    const int lr = lane & 15, hl = lane >> 4;
    int task = blockIdx.x * 4 + (threadIdx.x >> 5);      // 2048 tiles
    int m0 = (task >> 3) * 16;                           // 256 row tiles over B*NTOK
    int n0 = (task & 7) * 64;                            // 8 col groups over DIM

    floatx8 zf = {0.f,0.f,0.f,0.f,0.f,0.f,0.f,0.f};
    floatx8 acc[4];
    #pragma unroll
    for (int t = 0; t < 4; ++t) acc[t] = zf;

    const unsigned short* ar0 = attn_bf + (size_t)(m0 + lr) * INNER;
    const unsigned short* bp0 = WoT + (size_t)(n0 +  0 + lr) * INNER + hl * 16;
    const unsigned short* bp1 = WoT + (size_t)(n0 + 16 + lr) * INNER + hl * 16;
    const unsigned short* bp2 = WoT + (size_t)(n0 + 32 + lr) * INNER + hl * 16;
    const unsigned short* bp3 = WoT + (size_t)(n0 + 48 + lr) * INNER + hl * 16;

    AB aBuf[2], bBuf[2][4];
#define PROJ_LOAD(buf, ks) do {                                               \
        aBuf[buf].q[0] = *(const U4*)(ar0 + (ks)*32 +     hl * 16);           \
        aBuf[buf].q[1] = *(const U4*)(ar0 + (ks)*32 + 8 + hl * 16);           \
        bBuf[buf][0].q[0] = *(const U4*)(bp0 + (ks)*32);                      \
        bBuf[buf][0].q[1] = *(const U4*)(bp0 + (ks)*32 + 8);                  \
        bBuf[buf][1].q[0] = *(const U4*)(bp1 + (ks)*32);                      \
        bBuf[buf][1].q[1] = *(const U4*)(bp1 + (ks)*32 + 8);                  \
        bBuf[buf][2].q[0] = *(const U4*)(bp2 + (ks)*32);                      \
        bBuf[buf][2].q[1] = *(const U4*)(bp2 + (ks)*32 + 8);                  \
        bBuf[buf][3].q[0] = *(const U4*)(bp3 + (ks)*32);                      \
        bBuf[buf][3].q[1] = *(const U4*)(bp3 + (ks)*32 + 8);                  \
    } while (0)

    PROJ_LOAD(0, 0);
    #pragma unroll 2
    for (int ks = 0; ks < 32; ++ks) {
        int cur = ks & 1, nxt = cur ^ 1;
        if (ks < 31) PROJ_LOAD(nxt, ks + 1);             // prefetch next K-step
        #pragma unroll
        for (int t = 0; t < 4; ++t)
            acc[t] = WMMA_BF16(aBuf[cur], bBuf[cur][t], acc[t]);
    }
#undef PROJ_LOAD

    #pragma unroll
    for (int t = 0; t < 4; ++t) {
        float bias = bo[n0 + t * 16 + lr];
        #pragma unroll
        for (int r = 0; r < 8; ++r)
            proj[(size_t)(m0 + r + 8 * hl) * DIM + n0 + t * 16 + lr] = acc[t][r] + bias;
    }
}

// ---------------- gate: out = proj * sigmoid(proj @ gate_w + gate_b) ----------------

__global__ __launch_bounds__(256)
void k_gate(const float* __restrict__ proj, const float* __restrict__ gw,
            const float* __restrict__ gb, float* __restrict__ out) {
    __shared__ float red[256];
    int row = blockIdx.x;
    int t   = threadIdx.x;
    const float* pr = proj + (size_t)row * DIM;
    float p0 = pr[t], p1 = pr[t + 256];
    red[t] = p0 * gw[t] + p1 * gw[t + 256];
    __syncthreads();
    #pragma unroll
    for (int s = 128; s > 0; s >>= 1) {
        if (t < s) red[t] += red[t + s];
        __syncthreads();
    }
    float g = 1.0f / (1.0f + __expf(-(red[0] + gb[0])));
    out[(size_t)row * DIM + t]       = p0 * g;
    out[(size_t)row * DIM + t + 256] = p1 * g;
}

// ---------------- host launcher ----------------

extern "C" void kernel_launch(void* const* d_in, const int* in_sizes, int n_in,
                              void* d_out, int out_size, void* d_ws, size_t ws_size,
                              hipStream_t stream) {
    (void)in_sizes; (void)n_in; (void)out_size; (void)ws_size;
    const float* x     = (const float*)d_in[0];
    const float* y     = (const float*)d_in[1];
    const int*   tmask = (const int*)d_in[2];
    const float* Wq    = (const float*)d_in[3];
    const float* Wk    = (const float*)d_in[4];
    const float* Wv    = (const float*)d_in[5];
    const float* Wo    = (const float*)d_in[6];
    const float* bo    = (const float*)d_in[7];
    const float* pos   = (const float*)d_in[8];
    const float* gw    = (const float*)d_in[9];
    const float* gb    = (const float*)d_in[10];
    float* out = (float*)d_out;

    char* ws = (char*)d_ws;
    size_t off = 0;
    auto alloc = [&](size_t bytes) -> void* {
        void* p = ws + off;
        off += (bytes + 255) & ~(size_t)255;
        return p;
    };
    const size_t nEl = (size_t)B_ * NTOK * DIM;                     // 2,097,152
    unsigned short* xb      = (unsigned short*)alloc(nEl * 2);
    unsigned short* yb      = (unsigned short*)alloc(nEl * 2);
    unsigned short* WqT     = (unsigned short*)alloc((size_t)DIM * INNER * 2);
    unsigned short* WkT     = (unsigned short*)alloc((size_t)DIM * INNER * 2);
    unsigned short* WvT     = (unsigned short*)alloc((size_t)DIM * INNER * 2);
    unsigned short* WoT     = (unsigned short*)alloc((size_t)DIM * INNER * 2);
    unsigned short* qbuf    = (unsigned short*)alloc((size_t)B_ * INNER * NTOK * 2);
    unsigned short* kbuf    = (unsigned short*)alloc((size_t)B_ * INNER * NTOK * 2);
    unsigned short* vtbuf   = (unsigned short*)alloc((size_t)B_ * INNER * NTOK * 2);
    unsigned short* attn_bf = (unsigned short*)alloc((size_t)B_ * NTOK * INNER * 2);
    float*          proj    = (float*)alloc((size_t)B_ * NTOK * DIM * 4);

    // prep
    k_addpos_bf16<<<(nEl + 255) / 256, 256, 0, stream>>>(x, pos, xb, (int)nEl);
    k_addpos_bf16<<<(nEl + 255) / 256, 256, 0, stream>>>(y, pos, yb, (int)nEl);
    const int wEl = DIM * INNER;
    k_transpose_bf16<<<(wEl + 255) / 256, 256, 0, stream>>>(Wq, WqT, DIM, INNER);
    k_transpose_bf16<<<(wEl + 255) / 256, 256, 0, stream>>>(Wk, WkT, DIM, INNER);
    k_transpose_bf16<<<(wEl + 255) / 256, 256, 0, stream>>>(Wv, WvT, DIM, INNER);
    k_transpose_bf16<<<(wEl + 255) / 256, 256, 0, stream>>>(Wo, WoT, INNER, DIM);

    // QKV + RoPE: 3 mats * 2 batches * 64 row tiles (32 rows) * 16 heads = 6144 waves
    k_qkv<<<1536, 128, 0, stream>>>(xb, yb, WqT, WkT, WvT, qbuf, kbuf, vtbuf);

    // flash attention: B*H*128 q-tiles, single wave per block
    k_attn<<<4096, 32, 0, stream>>>(qbuf, kbuf, vtbuf, tmask, attn_bf);

    // output projection: 256 row tiles x 8 col groups / 4 waves per block
    k_proj<<<512, 128, 0, stream>>>(attn_bf, WoT, bo, proj);

    // gated output
    k_gate<<<(int)(B_ * NTOK), 256, 0, stream>>>(proj, gw, gb, out);
}
